// PreAttention_32006096289795
// MI455X (gfx1250) — matrix operand order repaired and checked by reading, not compiled, not run on hardware
//
#include <hip/hip_runtime.h>
#include <math.h>

typedef __attribute__((ext_vector_type(8)))  float        v8f;
typedef __attribute__((ext_vector_type(4)))  float        v4f;
typedef __attribute__((ext_vector_type(8)))  unsigned int v8u;
typedef __attribute__((ext_vector_type(4)))  unsigned int v4u;
typedef __attribute__((ext_vector_type(2)))  unsigned int v2u;
typedef __attribute__((ext_vector_type(16))) __bf16       v16bf;
typedef unsigned short bfu;

#define WDIM 64
#define SDIM 512
#define ODIM 512

// ---------- bf16 helpers (RNE) ----------
__device__ __forceinline__ bfu f2bf(float f) {
  unsigned u = __builtin_bit_cast(unsigned, f);
  u += 0x7FFFu + ((u >> 16) & 1u);
  return (bfu)(u >> 16);
}
__device__ __forceinline__ float bf2f(bfu h) {
  unsigned u = ((unsigned)h) << 16;
  return __builtin_bit_cast(float, u);
}
__device__ __forceinline__ unsigned pack2(float a, float b) {
  return (unsigned)f2bf(a) | ((unsigned)f2bf(b) << 16);
}

// ---------- WMMA fragment layouts (ISA 7.12.2, 16-bit A 16x32 / B 32x16) ----------
__device__ __forceinline__ int apack_idx(int row, int k, int nks) {
  int mt = row >> 4, l = row & 15;
  int ks = k >> 5, koff = k & 31;
  int lane = l | (((koff >> 3) & 1) << 4);
  int pos  = (koff & 7) | ((koff >> 4) << 3);
  return ((mt * nks + ks) << 9) + (lane << 4) + pos;
}
__device__ __forceinline__ int bpack_idx(int k, int n, int nnt) {
  int ks = k >> 5, kk = k & 31;
  int lane = (n & 15) | (((kk >> 3) & 1) << 4);
  int i  = (kk & 7) | ((kk >> 4) << 3);
  int nt = n >> 4;
  return ((ks * nnt + nt) << 9) + (lane << 4) + i;
}

__device__ __forceinline__ v16bf ld_frag(const bfu* tile, int lane) {
  v8u r = *reinterpret_cast<const v8u*>(tile + (lane << 4)); // 32 contiguous bytes/lane
  return __builtin_bit_cast(v16bf, r);
}
__device__ __forceinline__ v8f wmma_bf16(v16bf a, v16bf b, v8f c) {
  return __builtin_amdgcn_wmma_f32_16x16x32_bf16(false, a, false, b, (short)0, c, false, false);
}
__device__ __forceinline__ v8f zero8() {
  v8f z; for (int r = 0; r < 8; ++r) z[r] = 0.f; return z;
}

// ---------- Prep: f32 [K][N] weights -> bf16 B-fragment-packed ----------
__global__ void pack_b_kernel(const float* __restrict__ src, bfu* __restrict__ dst,
                              int K, int N) {
  int idx = blockIdx.x * blockDim.x + threadIdx.x;
  if (idx >= K * N) return;
  int k = idx / N, n = idx - k * N;
  dst[bpack_idx(k, n, N >> 4)] = f2bf(src[idx]);
}

// ---------- Spectral-norm power iteration: 1/sigma ----------
__global__ void sigma_kernel(const float* __restrict__ w, const float* __restrict__ u,
                             float* __restrict__ inv_sigma) {
  __shared__ float sv[512];
  __shared__ float su[512];
  __shared__ float red[512];
  int t = threadIdx.x;
  float acc = 0.f;                                    // v = W u
  for (int o = 0; o < 512; ++o) acc += w[t * 512 + o] * u[o];
  sv[t] = acc; red[t] = acc * acc;
  __syncthreads();
  for (int s = 256; s > 0; s >>= 1) { if (t < s) red[t] += red[t + s]; __syncthreads(); }
  float nv = sqrtf(red[0]) + 1e-12f;
  float vn = sv[t] / nv;
  __syncthreads();
  sv[t] = vn;
  __syncthreads();
  acc = 0.f;                                          // u2 = v W
  for (int s = 0; s < 512; ++s) acc += sv[s] * w[s * 512 + t];
  su[t] = acc; red[t] = acc * acc;
  __syncthreads();
  for (int s = 256; s > 0; s >>= 1) { if (t < s) red[t] += red[t + s]; __syncthreads(); }
  float nu = sqrtf(red[0]) + 1e-12f;
  float un = su[t] / nu;
  __syncthreads();
  su[t] = un;
  __syncthreads();
  acc = 0.f;                                          // sigma = v W u2
  for (int o = 0; o < 512; ++o) acc += w[t * 512 + o] * su[o];
  red[t] = sv[t] * acc;
  __syncthreads();
  for (int s = 256; s > 0; s >>= 1) { if (t < s) red[t] += red[t + s]; __syncthreads(); }
  if (t == 0) inv_sigma[0] = 1.0f / red[0];
}

// ---------- Fully fused per-(b,h) kernel ----------
__global__ __launch_bounds__(256, 1)
void fused_main(const float* __restrict__ x,
                const float* __restrict__ bf_v, const float* __restrict__ bg_v,
                const float* __restrict__ bh_v, const float* __restrict__ gamma_v,
                const float* __restrict__ ln1g, const float* __restrict__ ln1b,
                const float* __restrict__ ln2g, const float* __restrict__ ln2b,
                const bfu* __restrict__ wfp, const bfu* __restrict__ wgp,
                const bfu* __restrict__ whp, const bfu* __restrict__ wsnp,
                const float* __restrict__ inv_sigma,
                float* __restrict__ out) {
  extern __shared__ __align__(64) char smem[];
  bfu* xp = (bfu*)smem;           // [4mt][16ks] A-pack, 64KB  (reused as out1 A-pack)
  bfu* hp = xp + 32768;           // [2ks][32nt] B-pack, 64KB  (reused as y [64][512])
  bfu* fp = hp + 32768;           // f A-pack [4][2], 8KB
  bfu* gp = fp + 4096;            // g B-pack [2][4], 8KB
  bfu* ap = gp + 4096;            // att A-pack [4][2], 8KB
  float* st = (float*)(ap + 4096);// [0:64)=s1 sum [64:128)=s1 sq [128:192)=s2 sum [192:256)=s2 sq

  const int tid  = threadIdx.x;
  const int wave = tid >> 5;
  const int lane = tid & 31;
  const int batch = blockIdx.x;
  const float* xg = x   + (size_t)batch * (WDIM * SDIM);
  float*     outg = out + (size_t)batch * (WDIM * ODIM);

  // Phase 1: stage x -> LDS A-pack as bf16.  4 consecutive cols = 4 consecutive
  // u16 in one lane slot -> float4 global load + b64 LDS store.
  {
    const v4f* xg4 = reinterpret_cast<const v4f*>(xg);
    for (int c = tid; c < (WDIM * SDIM) / 4; c += 256) {
      v4f xv = xg4[c];
      int row  = c >> 7;
      int col0 = (c & 127) << 2;
      v2u pk;
      pk[0] = pack2(xv[0], xv[1]);
      pk[1] = pack2(xv[2], xv[3]);
      *reinterpret_cast<v2u*>(xp + apack_idx(row, col0, 16)) = pk;
    }
  }
  st[tid] = 0.f;
  __syncthreads();

  const int mt    = wave & 3;       // M-tile owned by this wave
  const int wh2   = wave >> 2;      // 0/1: N-tile parity
  const int lcol  = lane & 15;
  const int lrow8 = (lane >> 4) << 3;

  // Phase 2: f = x@wf+bf (waves 0-3), g = x@wg+bg (waves 4-7)   [64x512 * 512x64]
  {
    const bfu*   wp   = (wave < 4) ? wfp  : wgp;
    const float* bias = (wave < 4) ? bf_v : bg_v;
    for (int nt = 0; nt < 4; ++nt) {
      v8f acc = zero8();
      #pragma unroll 4
      for (int ks = 0; ks < 16; ++ks) {
        v16bf a = ld_frag(xp + ((mt * 16 + ks) << 9), lane);
        v16bf b = ld_frag(wp + ((ks * 4 + nt) << 9), lane);
        acc = wmma_bf16(a, b, acc);
      }
      int col = nt * 16 + lcol;
      float bv = bias[col];
      for (int r = 0; r < 8; ++r) {
        int row = mt * 16 + r + lrow8;
        float v = acc[r] + bv;
        if (wave < 4) fp[apack_idx(row, col, 2)] = f2bf(v);       // f: next A (M=w,K=c)
        else          gp[bpack_idx(col, row, 4)] = f2bf(v);       // g^T: next B (K=c,N=v)
      }
    }
  }
  __syncthreads();

  // Phase 3: att = sigmoid(f @ g^T)    [64x64 * 64x64]
  for (int j = 0; j < 2; ++j) {
    int nt = (wh2 << 1) + j;
    v8f acc = zero8();
    #pragma unroll
    for (int ks = 0; ks < 2; ++ks) {
      v16bf a = ld_frag(fp + ((mt * 2 + ks) << 9), lane);
      v16bf b = ld_frag(gp + ((ks * 4 + nt) << 9), lane);
      acc = wmma_bf16(a, b, acc);
    }
    int col = nt * 16 + lcol;
    for (int r = 0; r < 8; ++r) {
      int row = mt * 16 + r + lrow8;
      float s = 1.0f / (1.0f + __expf(-acc[r]));
      ap[apack_idx(row, col, 2)] = f2bf(s);                       // att: next A (M=w,K=v)
    }
  }
  __syncthreads();

  // Phase 4: h = x@wh + bh   [64x512 * 512x512], stored as B-pack (K=v, N=c).
  // In B-pack a lane's 8 accumulator rows are 8 consecutive u16 -> one b128 store.
  for (int j = 0; j < 16; ++j) {
    int nt = (j << 1) + wh2;
    v8f acc = zero8();
    #pragma unroll 8
    for (int ks = 0; ks < 16; ++ks) {
      v16bf a = ld_frag(xp + ((mt * 16 + ks) << 9), lane);
      v16bf b = ld_frag(whp + ((ks * 32 + nt) << 9), lane);
      if (ks < 15) __builtin_prefetch(whp + (((ks + 1) * 32 + nt) << 9) + (lane << 4), 0, 1);
      acc = wmma_bf16(a, b, acc);
    }
    int col = nt * 16 + lcol;
    float bv = bh_v[col];
    v4u hv;
    for (int jj = 0; jj < 4; ++jj)
      hv[jj] = pack2(acc[2 * jj] + bv, acc[2 * jj + 1] + bv);
    *reinterpret_cast<v4u*>(hp + bpack_idx(mt * 16 + lrow8, col, 32)) = hv;
  }
  __syncthreads();

  // Phase 5: attn_out = att@h; out1_pre = gamma*attn_out + x.
  // Row stats accumulate in registers across the whole j-loop (rows are
  // j-invariant per wave), reduced across lanes once at the end.
  {
    float gam = gamma_v[0];
    float ssum[8], ssq[8];
    for (int r = 0; r < 8; ++r) { ssum[r] = 0.f; ssq[r] = 0.f; }
    for (int j = 0; j < 16; ++j) {
      int nt = (j << 1) + wh2;
      v8f acc = zero8();
      #pragma unroll
      for (int ks = 0; ks < 2; ++ks) {
        v16bf a = ld_frag(ap + ((mt * 2 + ks) << 9), lane);
        v16bf b = ld_frag(hp + ((ks * 32 + nt) << 9), lane);
        acc = wmma_bf16(a, b, acc);
      }
      int col = nt * 16 + lcol;
      for (int r = 0; r < 8; ++r) {
        int row = mt * 16 + r + lrow8;
        float val = gam * acc[r] + xg[row * SDIM + col];
        xp[apack_idx(row, col, 16)] = f2bf(val);                  // out1 pre-LN, A-pack
        ssum[r] += val;
        ssq[r]  += val * val;
      }
    }
    for (int r = 0; r < 8; ++r) {
      float s = ssum[r], q = ssq[r];
      for (int msk = 1; msk < 16; msk <<= 1) {
        s += __shfl_xor(s, msk, 32);
        q += __shfl_xor(q, msk, 32);
      }
      if (lcol == 0) {
        int row = mt * 16 + r + lrow8;
        atomicAdd(&st[row], s); atomicAdd(&st[64 + row], q);
      }
    }
  }
  __syncthreads();

  // Phase 6: LayerNorm1 in-place on out1. 8 consecutive packed u16 share one row
  // and 8 consecutive columns -> 16B ds load/store per chunk, b128 param loads.
  for (int c = tid; c < (WDIM * SDIM) / 8; c += 256) {
    int ib   = c << 3;
    int tile = ib >> 9;
    int lid  = (ib >> 4) & 31;
    int pos0 = ib & 15;
    int row  = ((tile >> 4) << 4) + (lid & 15);
    int col0 = ((tile & 15) << 5) + ((pos0 >> 3) << 4) + ((lid >> 4) << 3);
    float m  = st[row]      * (1.f / 512.f);
    float mq = st[64 + row] * (1.f / 512.f);
    float rstd = rsqrtf(mq - m * m + 1e-6f);
    const v4f* g4 = reinterpret_cast<const v4f*>(ln1g + col0);
    const v4f* b4 = reinterpret_cast<const v4f*>(ln1b + col0);
    v4f g0 = g4[0], g1 = g4[1];
    v4f b0 = b4[0], b1 = b4[1];
    v4u d = *reinterpret_cast<v4u*>(xp + ib);
    v4u o;
    for (int jj = 0; jj < 4; ++jj) {
      float v0 = bf2f((bfu)(d[jj] & 0xFFFFu));
      float v1 = bf2f((bfu)(d[jj] >> 16));
      float ga = (jj < 2) ? g0[2 * jj]     : g1[2 * (jj - 2)];
      float gb = (jj < 2) ? g0[2 * jj + 1] : g1[2 * (jj - 2) + 1];
      float ba = (jj < 2) ? b0[2 * jj]     : b1[2 * (jj - 2)];
      float bb = (jj < 2) ? b0[2 * jj + 1] : b1[2 * (jj - 2) + 1];
      v0 = (v0 - m) * rstd * ga + ba;
      v1 = (v1 - m) * rstd * gb + bb;
      o[jj] = pack2(v0, v1);
    }
    *reinterpret_cast<v4u*>(xp + ib) = o;
  }
  __syncthreads();

  // Phase 7: y = relu(out1 @ (w_sn / sigma)); register-accumulated LN2 stats.
  {
    float isg = inv_sigma[0];
    float ssum[8], ssq[8];
    for (int r = 0; r < 8; ++r) { ssum[r] = 0.f; ssq[r] = 0.f; }
    for (int j = 0; j < 16; ++j) {
      int nt = (j << 1) + wh2;
      v8f acc = zero8();
      #pragma unroll 8
      for (int ks = 0; ks < 16; ++ks) {
        v16bf a = ld_frag(xp + ((mt * 16 + ks) << 9), lane);
        v16bf b = ld_frag(wsnp + ((ks * 32 + nt) << 9), lane);
        if (ks < 15) __builtin_prefetch(wsnp + (((ks + 1) * 32 + nt) << 9) + (lane << 4), 0, 1);
        acc = wmma_bf16(a, b, acc);
      }
      int col = nt * 16 + lcol;
      for (int r = 0; r < 8; ++r) {
        int row = mt * 16 + r + lrow8;
        float y = acc[r] * isg;
        y = y > 0.f ? y : 0.f;
        hp[row * ODIM + col] = f2bf(y);                           // plain [64][512]
        ssum[r] += y;
        ssq[r]  += y * y;
      }
    }
    for (int r = 0; r < 8; ++r) {
      float s = ssum[r], q = ssq[r];
      for (int msk = 1; msk < 16; msk <<= 1) {
        s += __shfl_xor(s, msk, 32);
        q += __shfl_xor(q, msk, 32);
      }
      if (lcol == 0) {
        int row = mt * 16 + r + lrow8;
        atomicAdd(&st[128 + row], s); atomicAdd(&st[192 + row], q);
      }
    }
  }
  __syncthreads();

  // Phase 8: LayerNorm2 + coalesced f32 output (2x b128 global store per chunk)
  for (int c = tid; c < (WDIM * ODIM) / 8; c += 256) {
    int ib   = c << 3;
    int row  = ib >> 9;
    int col0 = ib & 511;
    float m  = st[128 + row] * (1.f / 512.f);
    float mq = st[192 + row] * (1.f / 512.f);
    float rstd = rsqrtf(mq - m * m + 1e-6f);
    const v4f* g4 = reinterpret_cast<const v4f*>(ln2g + col0);
    const v4f* b4 = reinterpret_cast<const v4f*>(ln2b + col0);
    v4f g0 = g4[0], g1 = g4[1];
    v4f b0 = b4[0], b1 = b4[1];
    v4u d = *reinterpret_cast<const v4u*>(hp + ib);
    v4f r0, r1;
    for (int jj = 0; jj < 4; ++jj) {
      float v0 = bf2f((bfu)(d[jj] & 0xFFFFu));
      float v1 = bf2f((bfu)(d[jj] >> 16));
      float ga = (jj < 2) ? g0[2 * jj]     : g1[2 * (jj - 2)];
      float gb = (jj < 2) ? g0[2 * jj + 1] : g1[2 * (jj - 2) + 1];
      float ba = (jj < 2) ? b0[2 * jj]     : b1[2 * (jj - 2)];
      float bb = (jj < 2) ? b0[2 * jj + 1] : b1[2 * (jj - 2) + 1];
      v0 = (v0 - m) * rstd * ga + ba;
      v1 = (v1 - m) * rstd * gb + bb;
      if (jj < 2) { r0[2 * jj] = v0; r0[2 * jj + 1] = v1; }
      else        { r1[2 * (jj - 2)] = v0; r1[2 * (jj - 2) + 1] = v1; }
    }
    *reinterpret_cast<v4f*>(outg + ib)     = r0;
    *reinterpret_cast<v4f*>(outg + ib + 4) = r1;
  }
}

// ---------- host ----------
extern "C" void kernel_launch(void* const* d_in, const int* in_sizes, int n_in,
                              void* d_out, int out_size, void* d_ws, size_t ws_size,
                              hipStream_t stream) {
  (void)in_sizes; (void)n_in; (void)out_size; (void)ws_size;
  const float* x     = (const float*)d_in[0];
  const float* wf    = (const float*)d_in[1];
  const float* bf_v  = (const float*)d_in[2];
  const float* wg    = (const float*)d_in[3];
  const float* bg_v  = (const float*)d_in[4];
  const float* wh    = (const float*)d_in[5];
  const float* bh_v  = (const float*)d_in[6];
  const float* gam   = (const float*)d_in[7];
  const float* ln1g  = (const float*)d_in[8];
  const float* ln1b  = (const float*)d_in[9];
  const float* wsn   = (const float*)d_in[10];
  const float* usn   = (const float*)d_in[11];
  const float* ln2g  = (const float*)d_in[12];
  const float* ln2b  = (const float*)d_in[13];
  float* out = (float*)d_out;

  char* ws = (char*)d_ws;
  float* inv_sigma = (float*)ws;
  bfu* wfp  = (bfu*)(ws + 256);
  bfu* wgp  = (bfu*)(ws + 256 + 65536);
  bfu* whp  = (bfu*)(ws + 256 + 2 * 65536);
  bfu* wsnp = (bfu*)(ws + 256 + 2 * 65536 + 524288);

  pack_b_kernel<<<(512 * 64 + 255) / 256, 256, 0, stream>>>(wf, wfp, 512, 64);
  pack_b_kernel<<<(512 * 64 + 255) / 256, 256, 0, stream>>>(wg, wgp, 512, 64);
  pack_b_kernel<<<(512 * 512 + 255) / 256, 256, 0, stream>>>(wh, whp, 512, 512);
  pack_b_kernel<<<(512 * 512 + 255) / 256, 256, 0, stream>>>(wsn, wsnp, 512, 512);
  sigma_kernel<<<1, 512, 0, stream>>>(wsn, usn, inv_sigma);

  const int smem_bytes = (32768 + 32768 + 4096 + 4096 + 4096) * 2 + 256 * 4; // 156672
  fused_main<<<16 * 64, 256, smem_bytes, stream>>>(
      x, bf_v, bg_v, bh_v, gam, ln1g, ln1b, ln2g, ln2b,
      wfp, wgp, whp, wsnp, inv_sigma, out);
}
